// DeltaPredictor_86887188398358
// MI455X (gfx1250) — matrix operand order repaired and checked by reading, not compiled, not run on hardware
//
#include <hip/hip_runtime.h>
#include <hip/hip_bf16.h>
#include <math.h>

// ---------------------------------------------------------------------------
// DeltaPredictor for MI455X (gfx1250, wave32, WMMA).
//   encoder(conv+gelu) -> gates_x GEMM (bf16 WMMA, f32 acc, NT gx stream)
//   -> 126 fused GRU steps (async-LDS h panel + bf16 WMMA + gate math)
//   -> deconv+tanh+clip
// ---------------------------------------------------------------------------

typedef __bf16 bf16;
typedef __attribute__((ext_vector_type(16))) __bf16 v16bf;
typedef __attribute__((ext_vector_type(8)))  __bf16 v8bf;
typedef __attribute__((ext_vector_type(8)))  float  v8f;
typedef int v4i __attribute__((__vector_size__(16)));   // matches builtin param

#define B_   512
#define T_   128
#define T2_  126
#define CH_  32
#define HID_ 512
#define NG_  1536                 // 3*HID
#define M_   (B_ * T2_)           // 64512 rows of feat / gates_x

#if defined(__has_builtin)
#  if __has_builtin(__builtin_amdgcn_global_load_async_to_lds_b128)
#    define HAVE_ASYNC_LDS 1
#  endif
#endif

__device__ __forceinline__ void wait_asynccnt0() {
#if defined(__has_builtin)
#  if __has_builtin(__builtin_amdgcn_s_wait_asynccnt)
    __builtin_amdgcn_s_wait_asynccnt(0);
#  else
    asm volatile("s_wait_asynccnt 0x0" ::: "memory");
#  endif
#endif
}

// ---------------------------------------------------------------------------
// WMMA fragment loaders (CDNA5 16x16x32 bf16; same contiguous k order on A and B, not the ISA 7.12.2 table order):
//  A (16x32): lane l -> row (l&15); K = half*16 + {0..15} contiguous
//  B (32x16): lane l -> col (l&15); K = half*16 + {0..15} contiguous
// Both map to contiguous b128 loads from row-major bf16 storage
// (global_load_b128 or ds_load_b128 depending on the pointer's space).
// ---------------------------------------------------------------------------
__device__ __forceinline__ v16bf load_frag_a(const bf16* base, int ld,
                                             int row, int k0, int half) {
    union { v16bf v; v8bf h[2]; } u;
    const bf16* p = base + (size_t)row * ld + k0 + half * 16;
    u.h[0] = *(const v8bf*)(p);        // K = k0 + half*16 + 0..7
    u.h[1] = *(const v8bf*)(p + 8);    // K = k0 + half*16 + 8..15
    return u.v;
}

__device__ __forceinline__ v16bf load_frag_b(const bf16* base, int ld,
                                             int ncol, int k0, int half) {
    return *(const v16bf*)(base + (size_t)ncol * ld + k0 + half * 16);
}

__device__ __forceinline__ v8f wmma_bf16(v16bf a, v16bf b, v8f c) {
    return __builtin_amdgcn_wmma_f32_16x16x32_bf16(false, a, false, b,
                                                   (short)0, c, false, false);
}

__device__ __forceinline__ float sigmoidf_(float x) {
    return 1.0f / (1.0f + __expf(-x));
}

// ---------------------------------------------------------------------------
// Utility: f32 -> bf16 repack, zero-fill
// ---------------------------------------------------------------------------
__global__ void k_f32_to_bf16(const float* __restrict__ src, bf16* __restrict__ dst, long n) {
    long i = (long)blockIdx.x * blockDim.x + threadIdx.x;
    if (i < n) dst[i] = (bf16)src[i];
}

__global__ void k_zero_bf16(bf16* __restrict__ dst, long n) {
    long i = (long)blockIdx.x * blockDim.x + threadIdx.x;
    if (i < n) dst[i] = (bf16)0.0f;
}

// ---------------------------------------------------------------------------
// Encoder: diff-frames 4x4-patch conv (2ch -> 32ch) + exact GELU -> bf16 feat
// feat[m][o*16 + i*4 + j],  m = b*T2 + t  (curr = frame t+1, prev = frame t)
// ---------------------------------------------------------------------------
__global__ __launch_bounds__(256) void k_encoder(const float* __restrict__ frames,
                                                 const float* __restrict__ conv_w,
                                                 const float* __restrict__ conv_b,
                                                 bf16* __restrict__ feat, long total) {
    __shared__ float s_w[CH_ * 32];   // conv_w[o][c][p][q]
    __shared__ float s_b[CH_];
    for (int i = threadIdx.x; i < CH_ * 32; i += 256) s_w[i] = conv_w[i];
    if (threadIdx.x < CH_) s_b[threadIdx.x] = conv_b[threadIdx.x];
    __syncthreads();

    long idx = (long)blockIdx.x * 256 + threadIdx.x;
    if (idx >= total) return;
    int  f = (int)(idx & 511);
    long m = idx >> 9;
    int  o  = f >> 4;
    int  i4 = (f >> 2) & 3;
    int  j4 = f & 3;
    int  b  = (int)(m / T2_);
    int  t  = (int)(m % T2_);

    const float* cur = frames + ((size_t)b * T_ + t + 1) * 256;
    const float* prv = frames + ((size_t)b * T_ + t) * 256;
    const float* w   = s_w + o * 32;

    float s = s_b[o];
#pragma unroll
    for (int p = 0; p < 4; ++p) {
#pragma unroll
        for (int q = 0; q < 4; ++q) {
            int pix = (i4 * 4 + p) * 16 + j4 * 4 + q;
            s += cur[pix] * w[p * 4 + q] + prv[pix] * w[16 + p * 4 + q];
        }
    }
    // exact GELU: 0.5*x*(1+erf(x/sqrt(2)))
    float g = 0.5f * s * (1.0f + erff(s * 0.70710678118654752f));
    feat[idx] = (bf16)g;
}

// ---------------------------------------------------------------------------
// gates_x GEMM: gx[m][n] = feat[m][:] . w_ih[n][:] + b_ih[n]
// M=64512, N=1536, K=512. Block = 8 waves (2x4), block tile 64M x 128N,
// wave tile 32x32 = 4 WMMA tiles, K loop 16 x k32.
// gx is 396 MB (> 192 MB L2) streamed exactly once each way -> NT stores.
// ---------------------------------------------------------------------------
__global__ __launch_bounds__(256) void k_gates_gemm(const bf16* __restrict__ feat,
                                                    const bf16* __restrict__ wih,
                                                    const float* __restrict__ bih,
                                                    float* __restrict__ gx) {
    const int lane  = threadIdx.x & 31;
    const int wave  = threadIdx.x >> 5;
    const int waveM = wave & 1;        // 0..1
    const int waveN = wave >> 1;       // 0..3
    const int half  = lane >> 4;
    const int nlane = lane & 15;
    const int row0  = blockIdx.x * 64 + waveM * 32;
    const int col0  = blockIdx.y * 128 + waveN * 32;

    v8f acc[2][2];
    const v8f vzero = {0.f, 0.f, 0.f, 0.f, 0.f, 0.f, 0.f, 0.f};
#pragma unroll
    for (int i = 0; i < 2; ++i)
#pragma unroll
        for (int j = 0; j < 2; ++j) acc[i][j] = vzero;

#pragma unroll 4
    for (int kk = 0; kk < 16; ++kk) {
        const int k0 = kk * 32;
        // speculative prefetch of the next A panel (global_prefetch_b8)
        __builtin_prefetch(feat + (size_t)(row0 + nlane) * HID_ + k0 + 64, 0, 0);
        v16bf a0 = load_frag_a(feat, HID_, row0 + nlane,      k0, half);
        v16bf a1 = load_frag_a(feat, HID_, row0 + 16 + nlane, k0, half);
        v16bf b0 = load_frag_b(wih,  HID_, col0 + nlane,      k0, half);
        v16bf b1 = load_frag_b(wih,  HID_, col0 + 16 + nlane, k0, half);
        acc[0][0] = wmma_bf16(a0, b0, acc[0][0]);
        acc[0][1] = wmma_bf16(a0, b1, acc[0][1]);
        acc[1][0] = wmma_bf16(a1, b0, acc[1][0]);
        acc[1][1] = wmma_bf16(a1, b1, acc[1][1]);
    }

    // C/D layout: lane l, vgpr j -> (row = half*8 + j, col = l&15)
#pragma unroll
    for (int mt = 0; mt < 2; ++mt)
#pragma unroll
        for (int nt = 0; nt < 2; ++nt) {
            const int c = col0 + nt * 16 + nlane;
            const float bias = bih[c];
#pragma unroll
            for (int j = 0; j < 8; ++j) {
                const int r = row0 + mt * 16 + half * 8 + j;
                __builtin_nontemporal_store(acc[mt][nt][j] + bias,
                                            &gx[(size_t)r * NG_ + c]);
            }
        }
}

// ---------------------------------------------------------------------------
// One fused GRU step:  gh = h_{t-1} @ w_hh.T (+b_hh), then gate math, h_t out.
// Grid (16,8): block = 32 batch-rows x 64 hidden-cols of all 3 gates.
// The block's 32x512 bf16 h-panel (32 KB) is staged into LDS once via the
// CDNA5 async-copy path (ASYNCcnt) and shared by all 8 waves; WMMA A
// fragments then come from LDS (ds_load_b128), weights stream from L2.
// 8 waves (2x4): wave tile 16 rows x 16 cols x 3 gates = 3 WMMA accum tiles.
// ---------------------------------------------------------------------------
__global__ __launch_bounds__(256) void k_gru_step(const bf16* __restrict__ hprev,
                                                  const bf16* __restrict__ whh,
                                                  const float* __restrict__ bhh,
                                                  const float* __restrict__ gx,
                                                  bf16* __restrict__ hout, int t) {
    __shared__ bf16 s_h[32 * HID_];    // 32 KB: rows rowB..rowB+31, full K

    const int lane  = threadIdx.x & 31;
    const int wave  = threadIdx.x >> 5;
    const int waveM = wave & 1;        // 0..1
    const int waveN = wave >> 1;       // 0..3
    const int half  = lane >> 4;
    const int nlane = lane & 15;
    const int rowB  = blockIdx.x * 32;                // block batch-row base
    const int col0  = blockIdx.y * 64 + waveN * 16;   // hidden cols

    // ---- stage h panel: 32 KB = 2048 x 16B chunks, 8 per thread ----
    {
        const char* g = (const char*)(hprev + (size_t)rowB * HID_);
#if defined(HAVE_ASYNC_LDS)
        auto g1 = (__attribute__((address_space(1))) v4i*)(g);
        auto l3 = (__attribute__((address_space(3))) v4i*)(s_h);
#pragma unroll
        for (int i = 0; i < 8; ++i) {
            const int chunk = threadIdx.x + i * 256;   // 16-byte units
            __builtin_amdgcn_global_load_async_to_lds_b128(g1 + chunk, l3 + chunk, 0, 0);
        }
        wait_asynccnt0();
#else
#pragma unroll
        for (int i = 0; i < 8; ++i) {
            const int off = (threadIdx.x + i * 256) * 16;
            *(v8bf*)((char*)s_h + off) = *(const v8bf*)(g + off);
        }
#endif
        __syncthreads();
    }

    v8f acc[3];                        // [gate r/z/n], one 16x16 m-tile each
    const v8f vzero = {0.f, 0.f, 0.f, 0.f, 0.f, 0.f, 0.f, 0.f};
#pragma unroll
    for (int gidx = 0; gidx < 3; ++gidx) acc[gidx] = vzero;

#pragma unroll 2
    for (int kk = 0; kk < 16; ++kk) {
        const int k0 = kk * 32;
        v16bf a  = load_frag_a(s_h, HID_, waveM * 16 + nlane, k0, half);
        v16bf br = load_frag_b(whh, HID_, 0 * HID_ + col0 + nlane, k0, half);
        v16bf bz = load_frag_b(whh, HID_, 1 * HID_ + col0 + nlane, k0, half);
        v16bf bn = load_frag_b(whh, HID_, 2 * HID_ + col0 + nlane, k0, half);
        acc[0] = wmma_bf16(a, br, acc[0]);
        acc[1] = wmma_bf16(a, bz, acc[1]);
        acc[2] = wmma_bf16(a, bn, acc[2]);
    }

    const int   c   = col0 + nlane;
    const float bhr = bhh[c];
    const float bhz = bhh[HID_ + c];
    const float bhn = bhh[2 * HID_ + c];

#pragma unroll
    for (int j = 0; j < 8; ++j) {
        const int    lrow = waveM * 16 + half * 8 + j;  // row in LDS panel
        const int    b    = rowB + lrow;                // batch row
        const size_t gb   = ((size_t)b * T2_ + t) * NG_;
        const float ghr = acc[0][j] + bhr;
        const float ghz = acc[1][j] + bhz;
        const float ghn = acc[2][j] + bhn;
        // gx streamed once: non-temporal reads
        const float gxr = __builtin_nontemporal_load(&gx[gb + c]);
        const float gxz = __builtin_nontemporal_load(&gx[gb + HID_ + c]);
        const float gxn = __builtin_nontemporal_load(&gx[gb + 2 * HID_ + c]);
        const float r   = sigmoidf_(gxr + ghr);
        const float z   = sigmoidf_(gxz + ghz);
        const float n   = tanhf(gxn + r * ghn);
        const float hp  = (float)s_h[(size_t)lrow * HID_ + c];  // old h from LDS
        const float hn  = (1.0f - z) * n + z * hp;
        hout[(size_t)b * HID_ + c] = (bf16)hn;
    }
}

// ---------------------------------------------------------------------------
// Deconv + tanh delta + clip:  pred[b,t,pix] = clip(curr + tanh(h.dw + db))
// ---------------------------------------------------------------------------
__global__ __launch_bounds__(256) void k_deconv(const bf16* __restrict__ outs,
                                                const float* __restrict__ dw,
                                                const float* __restrict__ db,
                                                const float* __restrict__ frames,
                                                float* __restrict__ pred, long total) {
    __shared__ float s_w[CH_ * 16];
    for (int i = threadIdx.x; i < CH_ * 16; i += 256) s_w[i] = dw[i];
    __syncthreads();

    long idx = (long)blockIdx.x * 256 + threadIdx.x;
    if (idx >= total) return;
    int  pix = (int)(idx & 255);
    long bt  = idx >> 8;                 // b*T2 + t
    int  b   = (int)(bt / T2_);
    int  t   = (int)(bt % T2_);
    int  row = pix >> 4, colp = pix & 15;
    int  i4 = row >> 2, p = row & 3, j4 = colp >> 2, q = colp & 3;

    const bf16* h = outs + ((size_t)t * B_ + b) * HID_;
    float s = 0.0f;
#pragma unroll
    for (int c = 0; c < CH_; ++c)
        s += (float)h[c * 16 + i4 * 4 + j4] * s_w[c * 16 + p * 4 + q];

    float delta = tanhf(s + db[0]);
    float curr  = frames[((size_t)b * T_ + t + 1) * 256 + pix];
    float v = curr + delta;
    v = v < 0.0f ? 0.0f : (v > 1.0f ? 1.0f : v);
    __builtin_nontemporal_store(v, &pred[idx]);
}

// ---------------------------------------------------------------------------
// Host-side launch
// ---------------------------------------------------------------------------
extern "C" void kernel_launch(void* const* d_in, const int* in_sizes, int n_in,
                              void* d_out, int out_size, void* d_ws, size_t ws_size,
                              hipStream_t stream) {
    (void)in_sizes; (void)n_in; (void)out_size; (void)ws_size;
    const float* frames   = (const float*)d_in[0];
    const float* conv_w   = (const float*)d_in[1];
    const float* conv_b   = (const float*)d_in[2];
    const float* w_ih     = (const float*)d_in[3];
    const float* w_hh     = (const float*)d_in[4];
    const float* b_ih     = (const float*)d_in[5];
    const float* b_hh     = (const float*)d_in[6];
    const float* deconv_w = (const float*)d_in[7];
    const float* deconv_b = (const float*)d_in[8];
    float* pred = (float*)d_out;

    char*  ws  = (char*)d_ws;
    size_t off = 0;
    auto alloc = [&](size_t bytes) -> void* {
        void* p = ws + off;
        off = (off + bytes + 255) & ~(size_t)255;
        return p;
    };
    bf16*  feat   = (bf16*)alloc((size_t)M_ * HID_ * sizeof(bf16));   //  66 MB
    bf16*  wih_bf = (bf16*)alloc((size_t)NG_ * HID_ * sizeof(bf16));  // 1.5 MB
    bf16*  whh_bf = (bf16*)alloc((size_t)NG_ * HID_ * sizeof(bf16));  // 1.5 MB
    bf16*  hzero  = (bf16*)alloc((size_t)B_ * HID_ * sizeof(bf16));   // 0.5 MB
    bf16*  outs   = (bf16*)alloc((size_t)M_ * HID_ * sizeof(bf16));   //  66 MB
    float* gx     = (float*)alloc((size_t)M_ * NG_ * sizeof(float));  // 396 MB

    // 1) repack weights f32 -> bf16 (row-major layout already WMMA-friendly)
    long nw = (long)NG_ * HID_;
    k_f32_to_bf16<<<(nw + 255) / 256, 256, 0, stream>>>(w_ih, wih_bf, nw);
    k_f32_to_bf16<<<(nw + 255) / 256, 256, 0, stream>>>(w_hh, whh_bf, nw);
    long nz = (long)B_ * HID_;
    k_zero_bf16<<<(nz + 255) / 256, 256, 0, stream>>>(hzero, nz);

    // 2) encoder conv + GELU -> bf16 feat
    long ne = (long)M_ * HID_;
    k_encoder<<<(ne + 255) / 256, 256, 0, stream>>>(frames, conv_w, conv_b, feat, ne);

    // 3) gates_x = feat @ w_ih.T + b_ih   (M=64512, N=1536, K=512)
    dim3 gg(M_ / 64, NG_ / 128);
    k_gates_gemm<<<gg, 256, 0, stream>>>(feat, wih_bf, b_ih, gx);

    // 4) 126 sequential fused GRU steps (grid 16x8 = 128 WGs per step)
    for (int t = 0; t < T2_; ++t) {
        const bf16* hp = (t == 0) ? hzero : (outs + (size_t)(t - 1) * B_ * HID_);
        bf16*       ho = outs + (size_t)t * B_ * HID_;
        k_gru_step<<<dim3(16, 8), 256, 0, stream>>>(hp, whh_bf, b_hh, gx, ho, t);
    }

    // 5) deconv + tanh + clip
    long nd = (long)M_ * 256;
    k_deconv<<<(nd + 255) / 256, 256, 0, stream>>>(outs, deconv_w, deconv_b,
                                                   frames, pred, nd);
}
